// EmbedMatcher_8933531975988
// MI455X (gfx1250) — matrix operand order repaired and checked by reading, not compiled, not run on hardware
//
#include <hip/hip_runtime.h>

// ---------------------------------------------------------------------------
// EmbedMatcher for MI455X (gfx1250): all GEMMs via v_wmma_f32_16x16x32_bf16.
// Fragment loads use contiguous 16B chunks -> ds_load_b128 / global_load_b128.
// ---------------------------------------------------------------------------

typedef __attribute__((ext_vector_type(16))) __bf16 v16bf;
typedef __attribute__((ext_vector_type(8)))  float  v8f;

#define DIM     256
#define PAD_IDX 200000
#define BQ      4096
#define FEW     5
#define NBN     50
#define DMOD    512     // 2*D
#define DINN    1024    // d_inner
#define HSZ     1024    // LSTM hidden
#define FOURH   4096

union FragB { v16bf v; unsigned int u[8]; uint4 q[2]; };

__device__ __forceinline__ unsigned short f2bf(float f) {
  union { float f; unsigned int u; } x; x.f = f;
  unsigned int u = x.u;
  u += 0x7FFFu + ((u >> 16) & 1u);           // round-to-nearest-even
  return (unsigned short)(u >> 16);
}

// A-matrix fragment (16x32 bf16, MxK): lanes 0-15/16-31 = M rows; per ISA the
// lane's 16 elements are two contiguous 8-element runs at K = half*8 and 16+half*8.
__device__ __forceinline__ v16bf load_frag_a(const unsigned short* base, int row0, int ld, int k0) {
  int lane = threadIdx.x & 31;
  int r = row0 + (lane & 15);
  int half = lane >> 4;
  const unsigned short* p = base + (size_t)r * ld + k0 + (half << 3);
  FragB f;
  f.q[0] = *(const uint4*)p;         // K pairs (half*8+0..7)
  f.q[1] = *(const uint4*)(p + 16);  // K pairs (16+half*8+0..7)
  return f.v;
}

// B-matrix fragment (32x16 bf16, KxN) from row-major [N,K] storage:
// lanes = N columns; lane's 16 elements are two contiguous 8-element runs at K = half*8 and 16+half*8.
__device__ __forceinline__ v16bf load_frag_b(const unsigned short* base, int n0, int ld, int k0) {
  int lane = threadIdx.x & 31;
  int n = n0 + (lane & 15);
  int half = lane >> 4;
  const unsigned short* p = base + (size_t)n * ld + k0 + (half << 3);
  FragB f;
  f.q[0] = *(const uint4*)p;
  f.q[1] = *(const uint4*)(p + 16);
  return f.v;
}

__device__ __forceinline__ v8f wmma_bf16(v16bf a, v16bf b, v8f c) {
  return __builtin_amdgcn_wmma_f32_16x16x32_bf16(false, a, false, b, (short)0, c, false, false);
}

__device__ __forceinline__ float sigmoidf_(float x) { return 1.f / (1.f + __expf(-x)); }

// ---------------------------------------------------------------------------
__global__ void k_f32_to_bf16(const float* __restrict__ src, unsigned short* __restrict__ dst, int n) {
  int i = blockIdx.x * blockDim.x + threadIdx.x;
  if (i < n) dst[i] = f2bf(src[i]);
}

// ---------------------------------------------------------------------------
// Neighbor encoder: one block (256 thr / 8 waves) per entity (grid-strided).
// gcn_W cached in LDS as bf16 once per block.
__global__ __launch_bounds__(256)
void k_neighbor(const int* __restrict__ ids, int id_stride,
                const int* __restrict__ conns,          // [N, NB, 2]
                const float* __restrict__ emb,          // [NSYM+1, 256]
                const float* __restrict__ gcn_W,        // [256,256]
                const float* __restrict__ gcn_wb, const float* __restrict__ gcn_b,
                const float* __restrict__ gate_W,       // [1,512]
                const float* __restrict__ gate_wb, const float* __restrict__ gate_b,
                float* __restrict__ outv, int out_ld, int out_off,
                int N) {
  extern __shared__ char smem[];
  unsigned short* Wl    = (unsigned short*)smem;                 // 256*256 bf16 (128KB)
  unsigned short* P     = Wl + 256 * 256;                        // 64*256 bf16  (32KB)
  float* selfE  = (float*)(P + 64 * 256);                        // 256
  float* agg    = selfE + 256;                                   // 256
  float* simArr = agg + 256;                                     // 64
  float* wgt    = simArr + 64;                                   // 64
  float* red    = wgt + 64;                                      // 32

  const int tid = threadIdx.x;
  const int lane = tid & 31;
  const int w = tid >> 5;

  for (int i = tid; i < 256 * 256; i += 256) Wl[i] = f2bf(gcn_W[i]);
  __syncthreads();

  for (int e = blockIdx.x; e < N; e += gridDim.x) {
    int sid = ids[(size_t)e * id_stride];
    float sv = emb[(size_t)sid * DIM + tid];
    selfE[tid] = sv;
    agg[tid] = 0.f;
    if (tid < 64) simArr[tid] = -3.0e38f;
    for (int i = tid; i < 14 * 256; i += 256) P[50 * 256 + i] = 0; // pad rows 50..63

    float p2 = sv * sv;
    for (int o = 16; o; o >>= 1) p2 += __shfl_down(p2, o, 32);
    if (lane == 0) red[w] = p2;
    __syncthreads();
    float selfN;
    {
      float s = 0.f;
#pragma unroll
      for (int i = 0; i < 8; ++i) s += red[i];
      selfN = sqrtf(s);
    }

    // neighbors: wave w handles k = w, w+8, ...
    for (int k = w; k < NBN; k += 8) {
      int rel = conns[((size_t)e * NBN + k) * 2 + 0];
      int ent = conns[((size_t)e * NBN + k) * 2 + 1];
      const float* re = emb + (size_t)rel * DIM;
      const float* ee = emb + (size_t)ent * DIM;
      float dot = 0.f, nn = 0.f;
#pragma unroll
      for (int t = 0; t < 8; ++t) {
        int c = lane + (t << 5);
        float rv = re[c], ev = ee[c];
        P[k * 256 + c] = f2bf(rv * ev);
        dot += selfE[c] * ev;
        nn  += ev * ev;
      }
      for (int o = 16; o; o >>= 1) { dot += __shfl_down(dot, o, 32); nn += __shfl_down(nn, o, 32); }
      if (lane == 0) {
        float ne = sqrtf(nn);
        float denom = fmaxf(selfN, 1e-12f) * fmaxf(ne, 1e-12f);
        simArr[k] = (rel == PAD_IDX) ? -1e9f : (dot / denom);
      }
    }
    __syncthreads();

    // masked softmax over 50 (wave 0)
    if (w == 0) {
      float a = simArr[lane], b = simArr[lane + 32];
      float m = fmaxf(a, b);
      for (int o = 16; o; o >>= 1) m = fmaxf(m, __shfl_xor(m, o, 32));
      float ea = (lane      < NBN) ? __expf(a - m) : 0.f;
      float eb = (lane + 32 < NBN) ? __expf(b - m) : 0.f;
      float s = ea + eb;
      for (int o = 16; o; o >>= 1) s += __shfl_xor(s, o, 32);
      float inv = 1.f / s;
      wgt[lane] = ea * inv;
      wgt[lane + 32] = eb * inv;
    }
    __syncthreads();

    // msg = leaky_relu(P @ Wl^T + wb + b); agg = sum_m wgt[m]*msg[m,:]
    {
      const int row = lane & 15, half = lane >> 4;
      const int nt0 = w, nt1 = w + 8;
      float bias0 = gcn_wb[nt0 * 16 + row] + gcn_b[nt0 * 16 + row];
      float bias1 = gcn_wb[nt1 * 16 + row] + gcn_b[nt1 * 16 + row];
      v8f acc[4][2];
#pragma unroll
      for (int mt = 0; mt < 4; ++mt)
#pragma unroll
        for (int i = 0; i < 8; ++i) { acc[mt][0][i] = bias0; acc[mt][1][i] = bias1; }

      for (int k0 = 0; k0 < 256; k0 += 32) {
        v16bf b0 = load_frag_b(Wl, nt0 * 16, 256, k0);
        v16bf b1 = load_frag_b(Wl, nt1 * 16, 256, k0);
#pragma unroll
        for (int mt = 0; mt < 4; ++mt) {
          v16bf a = load_frag_a(P, mt * 16, 256, k0);
          acc[mt][0] = wmma_bf16(a, b0, acc[mt][0]);
          acc[mt][1] = wmma_bf16(a, b1, acc[mt][1]);
        }
      }
      float part0 = 0.f, part1 = 0.f;
#pragma unroll
      for (int mt = 0; mt < 4; ++mt)
#pragma unroll
        for (int v = 0; v < 8; ++v) {
          int m = mt * 16 + v + (half << 3);
          float wm = wgt[m];
          float x0 = acc[mt][0][v]; x0 = (x0 > 0.f) ? x0 : 0.01f * x0;
          float x1 = acc[mt][1][v]; x1 = (x1 > 0.f) ? x1 : 0.01f * x1;
          part0 += wm * x0; part1 += wm * x1;
        }
      atomicAdd(&agg[nt0 * 16 + row], part0);
      atomicAdd(&agg[nt1 * 16 + row], part1);
    }
    __syncthreads();

    // gate = sigmoid([self, agg] . gate_W + wb + b)
    float gp = gate_W[tid] * selfE[tid] + gate_W[256 + tid] * agg[tid];
    for (int o = 16; o; o >>= 1) gp += __shfl_down(gp, o, 32);
    if (lane == 0) red[w] = gp;
    __syncthreads();
    if (tid == 0) {
      float s = 0.f;
#pragma unroll
      for (int i = 0; i < 8; ++i) s += red[i];
      red[16] = sigmoidf_(s + gate_wb[0] + gate_b[0]);
    }
    __syncthreads();
    float gate = red[16];
    outv[(size_t)e * out_ld + out_off + tid] = tanhf(selfE[tid] + gate * agg[tid]);
    __syncthreads();
  }
}

// ---------------------------------------------------------------------------
// Support encoder: y = LN(x + (relu(x@W1^T+b1))@W2^T + b2). 16 rows per block.
__global__ __launch_bounds__(256)
void k_support_enc(const float* __restrict__ X, int M,
                   const unsigned short* __restrict__ W1, const float* __restrict__ b1,
                   const unsigned short* __restrict__ W2, const float* __restrict__ b2,
                   const float* __restrict__ lng, const float* __restrict__ lnb,
                   float* __restrict__ Y) {
  extern __shared__ char smem[];
  float* xs = (float*)smem;                                  // 16*512 f32
  unsigned short* xb = (unsigned short*)(xs + 16 * 512);     // 16*512 bf16
  unsigned short* hb = xb + 16 * 512;                        // 16*1024 bf16
  float* ys = (float*)(hb + 16 * 1024);                      // 16*512 f32

  const int tid = threadIdx.x, lane = tid & 31, w = tid >> 5;
  const int row = lane & 15, half = lane >> 4;
  const int m0 = blockIdx.x * 16;

  for (int i = tid; i < 16 * 512; i += 256) {
    int r = i >> 9, c = i & 511;
    float v = (m0 + r < M) ? X[(size_t)(m0 + r) * DMOD + c] : 0.f;
    xs[i] = v; xb[i] = f2bf(v);
  }
  __syncthreads();

  { // h = relu(x @ W1^T + b1) : [16,1024]
    v8f acc[8];
#pragma unroll
    for (int j = 0; j < 8; ++j) {
      float bv = b1[(w * 8 + j) * 16 + row];
#pragma unroll
      for (int i = 0; i < 8; ++i) acc[j][i] = bv;
    }
    for (int k0 = 0; k0 < 512; k0 += 32) {
      v16bf a = load_frag_a(xb, 0, 512, k0);
#pragma unroll
      for (int j = 0; j < 8; ++j) {
        v16bf b = load_frag_b(W1, (w * 8 + j) * 16, 512, k0);
        acc[j] = wmma_bf16(a, b, acc[j]);
      }
    }
#pragma unroll
    for (int j = 0; j < 8; ++j) {
      int n = (w * 8 + j) * 16 + row;
#pragma unroll
      for (int v = 0; v < 8; ++v)
        hb[(v + (half << 3)) * 1024 + n] = f2bf(fmaxf(acc[j][v], 0.f));
    }
  }
  __syncthreads();

  { // y = h @ W2^T + b2 + x : [16,512]
    v8f acc[4];
#pragma unroll
    for (int j = 0; j < 4; ++j) {
      float bv = b2[(w * 4 + j) * 16 + row];
#pragma unroll
      for (int i = 0; i < 8; ++i) acc[j][i] = bv;
    }
    for (int k0 = 0; k0 < 1024; k0 += 32) {
      v16bf a = load_frag_a(hb, 0, 1024, k0);
#pragma unroll
      for (int j = 0; j < 4; ++j) {
        v16bf b = load_frag_b(W2, (w * 4 + j) * 16, 1024, k0);
        acc[j] = wmma_bf16(a, b, acc[j]);
      }
    }
#pragma unroll
    for (int j = 0; j < 4; ++j) {
      int n = (w * 4 + j) * 16 + row;
#pragma unroll
      for (int v = 0; v < 8; ++v) {
        int m = v + (half << 3);
        ys[m * 512 + n] = acc[j][v] + xs[m * 512 + n];
      }
    }
  }
  __syncthreads();

  // layernorm (2 rows per wave)
  for (int r = w * 2; r < w * 2 + 2; ++r) {
    float s = 0.f, s2 = 0.f;
#pragma unroll
    for (int i = 0; i < 16; ++i) { float v = ys[r * 512 + lane + (i << 5)]; s += v; s2 += v * v; }
    for (int o = 16; o; o >>= 1) { s += __shfl_xor(s, o, 32); s2 += __shfl_xor(s2, o, 32); }
    float mu = s * (1.f / 512.f);
    float var = s2 * (1.f / 512.f) - mu * mu;
    float rstd = rsqrtf(var + 1e-5f);
    if (m0 + r < M) {
#pragma unroll
      for (int i = 0; i < 16; ++i) {
        int c = lane + (i << 5);
        Y[(size_t)(m0 + r) * DMOD + c] = lng[c] * (ys[r * 512 + c] - mu) * rstd + lnb[c];
      }
    }
  }
}

// ---------------------------------------------------------------------------
// Generic C[M,N] = A(f32,[M,K]) @ B(bf16,[N,K] row-major, leading dim ldb)^T.
// Block: 32 (M) x 128 (N); two 16x16 tiles per wave (B frag reused for 2 WMMAs).
// Requires K%128==0, N%128==0, M%32==0.
__global__ __launch_bounds__(256)
void k_gemm_bf16(const float* __restrict__ A, int lda,
                 const unsigned short* __restrict__ Bm, int ldb,
                 float* __restrict__ C, int ldc, int K) {
  extern __shared__ char smem[];
  unsigned short* Ab = (unsigned short*)smem;   // 32*128 bf16 (8KB)
  const int tid = threadIdx.x, lane = tid & 31, w = tid >> 5;
  const int row = lane & 15, half = lane >> 4;
  const int m0 = blockIdx.y * 32;
  const int n0 = (blockIdx.x * 8 + w) * 16;
  v8f acc0, acc1;
#pragma unroll
  for (int i = 0; i < 8; ++i) { acc0[i] = 0.f; acc1[i] = 0.f; }

  for (int k0 = 0; k0 < K; k0 += 128) {
    // prefetch next B chunk for this wave's N rows (global_prefetch_b8)
    if (k0 + 128 < K)
      __builtin_prefetch(Bm + (size_t)(n0 + row) * ldb + k0 + 128, 0, 0);
    __syncthreads();
    for (int i = tid; i < 32 * 128; i += 256) {
      int r = i >> 7, c = i & 127;
      Ab[i] = f2bf(A[(size_t)(m0 + r) * lda + k0 + c]);
    }
    __syncthreads();
#pragma unroll
    for (int kk = 0; kk < 128; kk += 32) {
      v16bf b  = load_frag_b(Bm, n0, ldb, k0 + kk);
      v16bf a0 = load_frag_a(Ab, 0, 128, kk);
      v16bf a1 = load_frag_a(Ab, 16, 128, kk);
      acc0 = wmma_bf16(a0, b, acc0);
      acc1 = wmma_bf16(a1, b, acc1);
    }
  }
#pragma unroll
  for (int v = 0; v < 8; ++v) {
    int m = v + (half << 3);
    C[(size_t)(m0 + m) * ldc + n0 + row]      = acc0[v];
    C[(size_t)(m0 + 16 + m) * ldc + n0 + row] = acc1[v];
  }
}

// ---------------------------------------------------------------------------
__global__ void k_support_mean(const float* __restrict__ senc, float* __restrict__ sg) {
  int i = blockIdx.x * blockDim.x + threadIdx.x;
  if (i < DMOD) {
    float s = 0.f;
#pragma unroll
    for (int r = 0; r < FEW; ++r) s += senc[r * DMOD + i];
    sg[i] = s * (1.f / FEW);
  }
}

// gates_sup[j] = sum_k whh[j, 512+k] * sg[k]
__global__ void k_gates_sup(const float* __restrict__ whh, const float* __restrict__ sg,
                            float* __restrict__ outp) {
  __shared__ float sgl[DMOD];
  for (int i = threadIdx.x; i < DMOD; i += blockDim.x) sgl[i] = sg[i];
  __syncthreads();
  int j = blockIdx.x * blockDim.x + threadIdx.x;
  if (j < FOURH) {
    const float* r = whh + (size_t)j * HSZ + DMOD;
    float s = 0.f;
    for (int k = 0; k < DMOD; ++k) s += r[k] * sgl[k];
    outp[j] = s;
  }
}

// fused LSTM pointwise step
__global__ void k_lstm_step(const float* __restrict__ wq,
                            const float* __restrict__ gh,     // null at step 0
                            const float* __restrict__ bih, const float* __restrict__ bhh,
                            const float* __restrict__ sup,    // null at step 0
                            float* __restrict__ c, int first,
                            const float* __restrict__ qenc,
                            float* __restrict__ h) {
  int idx = blockIdx.x * blockDim.x + threadIdx.x;
  if (idx >= BQ * HSZ) return;
  int rowi = idx >> 10, hc = idx & 1023;
  size_t base = (size_t)rowi * FOURH;
  float g4[4];
#pragma unroll
  for (int q = 0; q < 4; ++q) {
    int j = q * 1024 + hc;
    float g = wq[base + j] + bih[j] + bhh[j];
    if (sup) g += sup[j];
    if (gh)  g += gh[base + j];
    g4[q] = g;
  }
  float cp = first ? 0.f : c[idx];
  float c2 = sigmoidf_(g4[1]) * cp + sigmoidf_(g4[0]) * tanhf(g4[2]);
  c[idx] = c2;
  float hf = sigmoidf_(g4[3]) * tanhf(c2);
  if (hc < DMOD) h[(size_t)rowi * DMOD + hc] = qenc[(size_t)rowi * DMOD + hc] + hf;
}

__global__ void k_scores(const float* __restrict__ h, const float* __restrict__ sg,
                         float* __restrict__ outp) {
  int lane = threadIdx.x & 31, w = threadIdx.x >> 5;
  int row = blockIdx.x * 8 + w;
  if (row >= BQ) return;
  float s = 0.f;
#pragma unroll
  for (int i = 0; i < 16; ++i) { int c = lane + (i << 5); s += h[(size_t)row * DMOD + c] * sg[c]; }
  for (int o = 16; o; o >>= 1) s += __shfl_down(s, o, 32);
  if (lane == 0) outp[row] = s;
}

// ---------------------------------------------------------------------------
extern "C" void kernel_launch(void* const* d_in, const int* in_sizes, int n_in,
                              void* d_out, int out_size, void* d_ws, size_t ws_size,
                              hipStream_t stream) {
  const int*   query   = (const int*)d_in[0];
  const int*   support = (const int*)d_in[1];
  const int*   q_l1    = (const int*)d_in[2];
  const int*   q_r1    = (const int*)d_in[4];
  const int*   s_l1    = (const int*)d_in[6];
  const int*   s_r1    = (const int*)d_in[8];
  const float* emb     = (const float*)d_in[10];
  const float* gcn_W   = (const float*)d_in[11];
  const float* gcn_wb  = (const float*)d_in[12];
  const float* gcn_b   = (const float*)d_in[13];
  const float* gate_W  = (const float*)d_in[14];
  const float* gate_wb = (const float*)d_in[15];
  const float* gate_b  = (const float*)d_in[16];
  const float* se_W1   = (const float*)d_in[17];
  const float* se_b1   = (const float*)d_in[18];
  const float* se_W2   = (const float*)d_in[19];
  const float* se_b2   = (const float*)d_in[20];
  const float* ln_g    = (const float*)d_in[21];
  const float* ln_b    = (const float*)d_in[22];
  const float* wih     = (const float*)d_in[23];
  const float* whh     = (const float*)d_in[24];
  const float* bih     = (const float*)d_in[25];
  const float* bhh     = (const float*)d_in[26];
  float* out = (float*)d_out;

  // workspace carve-up
  char* ws = (char*)d_ws;
  size_t off = 0;
  auto carve = [&](size_t bytes) { char* p = ws + off; off = (off + bytes + 255) & ~(size_t)255; return p; };
  unsigned short* w1bf  = (unsigned short*)carve((size_t)DINN * DMOD * 2);
  unsigned short* w2bf  = (unsigned short*)carve((size_t)DMOD * DINN * 2);
  unsigned short* wihbf = (unsigned short*)carve((size_t)FOURH * DMOD * 2);
  unsigned short* whhbf = (unsigned short*)carve((size_t)FOURH * HSZ * 2);
  float* qvec = (float*)carve((size_t)BQ * DMOD * 4);
  float* svec = (float*)carve((size_t)16 * DMOD * 4);
  float* qenc = (float*)carve((size_t)BQ * DMOD * 4);
  float* senc = (float*)carve((size_t)16 * DMOD * 4);
  float* sg   = (float*)carve((size_t)DMOD * 4);
  float* gsup = (float*)carve((size_t)FOURH * 4);
  float* wq   = (float*)carve((size_t)BQ * FOURH * 4);
  float* ghx  = (float*)carve((size_t)BQ * FOURH * 4);
  float* cbuf = (float*)carve((size_t)BQ * HSZ * 4);
  float* hbuf = (float*)carve((size_t)BQ * DMOD * 4);
  (void)ws_size; (void)in_sizes; (void)n_in; (void)out_size;

  // 0) weight conversion to bf16
  k_f32_to_bf16<<<(DINN * DMOD + 255) / 256, 256, 0, stream>>>(se_W1, w1bf, DINN * DMOD);
  k_f32_to_bf16<<<(DMOD * DINN + 255) / 256, 256, 0, stream>>>(se_W2, w2bf, DMOD * DINN);
  k_f32_to_bf16<<<(FOURH * DMOD + 255) / 256, 256, 0, stream>>>(wih, wihbf, FOURH * DMOD);
  k_f32_to_bf16<<<(FOURH * HSZ + 255) / 256, 256, 0, stream>>>(whh, whhbf, FOURH * HSZ);

  // 1) neighbor encoders
  size_t nb_smem = 256 * 256 * 2 + 64 * 256 * 2 + (256 + 256 + 64 + 64 + 32) * sizeof(float);
  k_neighbor<<<256, 256, nb_smem, stream>>>(query + 0, 2, q_l1, emb, gcn_W, gcn_wb, gcn_b,
                                            gate_W, gate_wb, gate_b, qvec, DMOD, 0, BQ);
  k_neighbor<<<256, 256, nb_smem, stream>>>(query + 1, 2, q_r1, emb, gcn_W, gcn_wb, gcn_b,
                                            gate_W, gate_wb, gate_b, qvec, DMOD, DIM, BQ);
  k_neighbor<<<FEW, 256, nb_smem, stream>>>(support + 0, 2, s_l1, emb, gcn_W, gcn_wb, gcn_b,
                                            gate_W, gate_wb, gate_b, svec, DMOD, 0, FEW);
  k_neighbor<<<FEW, 256, nb_smem, stream>>>(support + 1, 2, s_r1, emb, gcn_W, gcn_wb, gcn_b,
                                            gate_W, gate_wb, gate_b, svec, DMOD, DIM, FEW);

  // 2) support encoder on query_vec and support_vec
  size_t se_smem = 16 * DMOD * 4 + 16 * DMOD * 2 + 16 * DINN * 2 + 16 * DMOD * 4;
  k_support_enc<<<BQ / 16, 256, se_smem, stream>>>(qvec, BQ, w1bf, se_b1, w2bf, se_b2, ln_g, ln_b, qenc);
  k_support_enc<<<1, 256, se_smem, stream>>>(svec, FEW, w1bf, se_b1, w2bf, se_b2, ln_g, ln_b, senc);

  // 3) support_g = mean; constant support contribution to recurrent gates
  k_support_mean<<<2, 256, 0, stream>>>(senc, sg);
  k_gates_sup<<<FOURH / 256, 256, 0, stream>>>(whh, sg, gsup);

  // 4) loop-invariant wih_q = qenc @ wih^T   [4096 x 4096]
  k_gemm_bf16<<<dim3(FOURH / 128, BQ / 32), 256, 32 * 128 * 2, stream>>>(
      qenc, DMOD, wihbf, DMOD, wq, FOURH, DMOD);

  // 5) 4 LSTM process steps (attention over single support vector == identity)
  for (int s = 0; s < 4; ++s) {
    if (s > 0)
      k_gemm_bf16<<<dim3(FOURH / 128, BQ / 32), 256, 32 * 128 * 2, stream>>>(
          hbuf, DMOD, whhbf, HSZ, ghx, FOURH, DMOD);   // h @ whh[:, :512]^T
    k_lstm_step<<<(BQ * HSZ) / 256, 256, 0, stream>>>(
        wq, (s > 0) ? ghx : nullptr, bih, bhh, (s > 0) ? gsup : nullptr,
        cbuf, (s == 0) ? 1 : 0, qenc, hbuf);
  }

  // 6) scores = h_last . support_g
  k_scores<<<BQ / 8, 256, 0, stream>>>(hbuf, sg, out);
}
